// SegmentAwarePositionalEncoding_70772471104043
// MI455X (gfx1250) — compile-verified, matrix operand order checked by reading, not run on hardware
//
#include <hip/hip_runtime.h>
#include <hip/hip_bf16.h>

// ---------------------------------------------------------------------------
// SegmentAwarePositionalEncoding for MI455X (gfx1250, wave32, WMMA)
//
// pe = base@W_top  (batch-independent, bf16x3 split WMMA GEMM, computed once)
//    + feats@(seg_W@W_bot) + (seg_b@W_bot + proj_b)   (rank-3 epilogue FMA)
//    + codon/motif sin terms
// ---------------------------------------------------------------------------

#define D_MODEL 768
#define MAX_LEN 32768
#define BATCH   8
#define SEQ     8192
#define KTOT    768                 // inner dim of base GEMM
#define KBLK    32                  // K per bf16 WMMA
#define NKB     (KTOT / KBLK)       // 24 K-blocks
#define MTILES  (SEQ / 16)          // 512
#define NTILES  (D_MODEL / 16)      // 48
#define TWOPI   6.28318530717958647692f

typedef __bf16 bf16_t;
typedef bf16_t v16bf __attribute__((ext_vector_type(16)));
typedef float  v8f   __attribute__((ext_vector_type(8)));

// Motif frequencies: for period p in {8,10,21,147}: e in {0,2,4,6}, f = e*2pi/p
__device__ __constant__ float kMotifFreq[16] = {
    0.0f, 2.0f * TWOPI / 8.0f,   4.0f * TWOPI / 8.0f,   6.0f * TWOPI / 8.0f,
    0.0f, 2.0f * TWOPI / 10.0f,  4.0f * TWOPI / 10.0f,  6.0f * TWOPI / 10.0f,
    0.0f, 2.0f * TWOPI / 21.0f,  4.0f * TWOPI / 21.0f,  6.0f * TWOPI / 21.0f,
    0.0f, 2.0f * TWOPI / 147.0f, 4.0f * TWOPI / 147.0f, 6.0f * TWOPI / 147.0f};

// CDNA5 16-bit A/B 16x32 VGPR layout (ISA 7.12.2): element i (0..15) of a
// lane's v16bf maps to K: VGPR v=i/2 half h=i&1 -> K = 2v+h (v<4) else
// 16+2(v-4)+h, plus 8 if lane>=16.  Row/col = lane & 15.
__device__ __forceinline__ int wmma_ab_k(int lane, int i) {
  int v = i >> 1, h = i & 1;
  int k = (v < 4) ? (2 * v + h) : (16 + 2 * (v - 4) + h);
  return k + ((lane & 16) ? 8 : 0);
}

// ---- pack sinusoidal base PE directly into WMMA-A layout (bf16 hi + lo) ----
__global__ void sapk_pack_base(bf16_t* __restrict__ Ahi, bf16_t* __restrict__ Alo) {
  int e = blockIdx.x * blockDim.x + threadIdx.x;   // SEQ*KTOT elements
  int i     = e & 15;
  int idx16 = e >> 4;
  int lane  = idx16 & 31;
  int t     = idx16 >> 5;
  int kb    = t % NKB;
  int mt    = t / NKB;
  int s = mt * 16 + (lane & 15);
  int k = kb * KBLK + wmma_ab_k(lane, i);
  int j = k >> 1;
  // div = exp(2j * (-ln(10000)/768)); base[s,2j]=sin(s*div), base[s,2j+1]=cos
  float div = expf((float)(2 * j) * (-9.210340371976184f / (float)D_MODEL));
  float ang = (float)s * div;
  float val = (k & 1) ? cosf(ang) : sinf(ang);
  bf16_t hi = (bf16_t)val;
  bf16_t lo = (bf16_t)(val - (float)hi);
  Ahi[e] = hi;
  Alo[e] = lo;
}

// ---- pack proj_W[0:768, :] into WMMA-B layout (bf16 hi + lo) ----
__global__ void sapk_pack_w(const float* __restrict__ proj_W,
                            bf16_t* __restrict__ Bhi, bf16_t* __restrict__ Blo) {
  int e = blockIdx.x * blockDim.x + threadIdx.x;   // KTOT*D_MODEL elements
  int i     = e & 15;
  int idx16 = e >> 4;
  int lane  = idx16 & 31;
  int t     = idx16 >> 5;
  int kb    = t % NKB;
  int nt    = t / NKB;
  int d = nt * 16 + (lane & 15);
  int k = kb * KBLK + wmma_ab_k(lane, i);
  float val = proj_W[(size_t)k * D_MODEL + d];
  bf16_t hi = (bf16_t)val;
  bf16_t lo = (bf16_t)(val - (float)hi);
  Bhi[e] = hi;
  Blo[e] = lo;
}

// ---- fold seg_W/seg_b through proj_W[768:832] : M[3][768], c[768] ----
__global__ void sapk_prep_mc(const float* __restrict__ seg_W, const float* __restrict__ seg_b,
                             const float* __restrict__ proj_W, const float* __restrict__ proj_b,
                             float* __restrict__ Mc) {
  int d = blockIdx.x * blockDim.x + threadIdx.x;
  if (d >= D_MODEL) return;
  float m0 = 0.f, m1 = 0.f, m2 = 0.f, c = 0.f;
  for (int t = 0; t < 64; ++t) {
    float wv = proj_W[(size_t)(D_MODEL + t) * D_MODEL + d];
    m0 += seg_W[t]       * wv;
    m1 += seg_W[64 + t]  * wv;
    m2 += seg_W[128 + t] * wv;
    c  += seg_b[t]       * wv;
  }
  Mc[d]            = m0;
  Mc[768 + d]      = m1;
  Mc[2 * 768 + d]  = m2;
  Mc[3 * 768 + d]  = c + proj_b[d];
}

// ---- wave32 ballot scan: segment start (cummax fwd) + next start (bwd) ----
// One wave per batch row. feats = {pos_in_patch, len_norm, gpos, raw_pos}
__global__ void sapk_segscan(const int* __restrict__ pb, const int* __restrict__ op,
                             int* __restrict__ start_pos, float4* __restrict__ feats) {
  const int b = blockIdx.x;
  const int lane = threadIdx.x;
  const int* pbr = pb + (size_t)b * SEQ;
  const int* opr = op + (size_t)b * SEQ;
  int*    spr = start_pos + (size_t)b * SEQ;
  float4* fr  = feats + (size_t)b * SEQ;

  unsigned long long lmaskp1 = (((unsigned long long)2 << lane) - 1ull); // bits 0..lane

  int carry = 0;  // most recent segment start (p==0 is always a start)
  for (int c = 0; c < SEQ / 32; ++c) {
    int p = c * 32 + lane;
    bool st = (pbr[p] != 0) || (p == 0);
    unsigned int bal   = (unsigned int)__ballot(st);
    unsigned int below = bal & (unsigned int)lmaskp1;
    int sp = below ? (c * 32 + (31 - __builtin_clz(below))) : carry;
    spr[p] = sp;
    carry = __shfl(sp, 31);
  }

  int carryN = SEQ;  // next segment start strictly after this chunk
  for (int c = SEQ / 32 - 1; c >= 0; --c) {
    int p = c * 32 + lane;
    int sp = spr[p];
    bool st = (sp == p);
    unsigned int bal   = (unsigned int)__ballot(st);
    unsigned int above = bal & (unsigned int)(~lmaskp1);   // bits lane+1..31
    int ns = above ? (c * 32 + __builtin_ctz(above)) : carryN;
    float length = (float)(ns - sp);
    float pos_in = (float)(p - sp) / fmaxf(length - 1.0f, 1.0f);
    float posf   = (float)opr[p];
    float4 f;
    f.x = pos_in;
    f.y = length / (float)SEQ;
    f.z = posf / (float)MAX_LEN;
    f.w = posf;
    fr[p] = f;
    carryN = bal ? (c * 32 + __builtin_ctz(bal)) : carryN;  // ballot is wave-uniform
  }
}

// ---- bf16x3 split-precision WMMA GEMM: baseProj[S,768] = base @ W_top -----
// x*w ~= hi*whi + hi*wlo + lo*whi, fp32 accumulate -> ~fp32 accuracy at bf16
// WMMA rates. One 16x16 tile per wave; 8 waves/block share the A row panel.
__global__ void sapk_gemm(const v16bf* __restrict__ Ahi, const v16bf* __restrict__ Alo,
                          const v16bf* __restrict__ Bhi, const v16bf* __restrict__ Blo,
                          float* __restrict__ out) {
  const int lane = threadIdx.x & 31;
  const int wave = threadIdx.x >> 5;
  const int mt = blockIdx.x;                // 0..511
  const int nt = blockIdx.y * 8 + wave;     // 0..47

  v8f acc = {0.f, 0.f, 0.f, 0.f, 0.f, 0.f, 0.f, 0.f};
  const int aBase = (mt * NKB) * 32 + lane;
  const int bBase = (nt * NKB) * 32 + lane;

  for (int kb = 0; kb < NKB; ++kb) {
    v16bf ah = Ahi[aBase + kb * 32];
    v16bf al = Alo[aBase + kb * 32];
    v16bf bh = Bhi[bBase + kb * 32];
    v16bf bl = Blo[bBase + kb * 32];
    if (kb + 1 < NKB) {
      __builtin_prefetch((const void*)&Ahi[aBase + (kb + 1) * 32], 0, 0);
      __builtin_prefetch((const void*)&Bhi[bBase + (kb + 1) * 32], 0, 0);
    }
    acc = __builtin_amdgcn_wmma_f32_16x16x32_bf16(false, ah, false, bh, (short)0, acc, false, false);
    acc = __builtin_amdgcn_wmma_f32_16x16x32_bf16(false, ah, false, bl, (short)0, acc, false, false);
    acc = __builtin_amdgcn_wmma_f32_16x16x32_bf16(false, al, false, bh, (short)0, acc, false, false);
  }

  // C/D layout: VGPR r -> M = r (+8 for lanes 16..31), N = lane & 15
  const int row0 = mt * 16 + ((lane & 16) ? 8 : 0);
  const int col  = nt * 16 + (lane & 15);
#pragma unroll
  for (int r = 0; r < 8; ++r)
    out[(size_t)(row0 + r) * D_MODEL + col] = acc[r];
}

// ---- store-bound fused epilogue: rank-3 term + bias + codon/motif sins ----
__global__ void sapk_fuse(const float4* __restrict__ baseProj, const float4* __restrict__ feats,
                          const float4* __restrict__ Mc, float4* __restrict__ out) {
  int idx  = blockIdx.x * blockDim.x + threadIdx.x;  // BATCH*SEQ*192
  int d4   = idx % 192;
  int rest = idx / 192;                               // b*SEQ + s
  int s    = rest & (SEQ - 1);

  float4 bp = baseProj[(size_t)s * 192 + d4];
  float4 f  = feats[rest];
  float4 m0 = Mc[d4], m1 = Mc[192 + d4], m2 = Mc[384 + d4], cc = Mc[576 + d4];

  float4 o;
  o.x = bp.x + f.x * m0.x + f.y * m1.x + f.z * m2.x + cc.x;
  o.y = bp.y + f.x * m0.y + f.y * m1.y + f.z * m2.y + cc.y;
  o.z = bp.z + f.x * m0.z + f.y * m1.z + f.z * m2.z + cc.z;
  o.w = bp.w + f.x * m0.w + f.y * m1.w + f.z * m2.w + cc.w;

  const float pos = f.w;
  const int d = d4 * 4;
  float* oc = &o.x;
  if (d4 < 24) {  // columns 0..95: codon freqs f(col) = 2*col * 2pi/3
    const float w0 = TWOPI / 3.0f;
#pragma unroll
    for (int c = 0; c < 4; ++c)
      oc[c] += 0.1f * sinf(pos * ((float)(2 * (d + c)) * w0));
  }
  if (d4 >= 188) {  // columns 752..767: motif freqs
#pragma unroll
    for (int c = 0; c < 4; ++c)
      oc[c] += 0.05f * sinf(pos * kMotifFreq[d + c - 752]);
  }
  out[idx] = o;
}

extern "C" void kernel_launch(void* const* d_in, const int* in_sizes, int n_in,
                              void* d_out, int out_size, void* d_ws, size_t ws_size,
                              hipStream_t stream) {
  (void)in_sizes; (void)n_in; (void)out_size; (void)ws_size;
  // inputs: [0]=seq_len (unused, fixed 8192), [1]=patch_boundaries i32[8,8192],
  // [2]=original_positions i32[8,8192], [3]=seg_W f32[3,64], [4]=seg_b f32[64],
  // [5]=proj_W f32[832,768], [6]=proj_b f32[768]
  const int*   pb     = (const int*)d_in[1];
  const int*   op     = (const int*)d_in[2];
  const float* seg_W  = (const float*)d_in[3];
  const float* seg_b  = (const float*)d_in[4];
  const float* proj_W = (const float*)d_in[5];
  const float* proj_b = (const float*)d_in[6];

  char* w = (char*)d_ws;
  const size_t szA  = (size_t)SEQ * KTOT * sizeof(bf16_t);      // 12.6 MB
  const size_t szB  = (size_t)KTOT * D_MODEL * sizeof(bf16_t);  // 1.18 MB
  const size_t szBP = (size_t)SEQ * D_MODEL * sizeof(float);    // 25.2 MB
  bf16_t* Ahi      = (bf16_t*)(w);
  bf16_t* Alo      = (bf16_t*)(w + szA);
  bf16_t* Bhi      = (bf16_t*)(w + 2 * szA);
  bf16_t* Blo      = (bf16_t*)(w + 2 * szA + szB);
  float*  baseProj = (float*) (w + 2 * szA + 2 * szB);
  float*  Mc       = (float*) (w + 2 * szA + 2 * szB + szBP);
  float4* feats    = (float4*)(w + 2 * szA + 2 * szB + szBP + 4 * 768 * sizeof(float));
  int*    startPos = (int*)   (w + 2 * szA + 2 * szB + szBP + 4 * 768 * sizeof(float)
                                 + (size_t)BATCH * SEQ * sizeof(float4));

  sapk_pack_base<<<(SEQ * KTOT) / 256, 256, 0, stream>>>(Ahi, Alo);
  sapk_pack_w  <<<(KTOT * D_MODEL) / 256, 256, 0, stream>>>(proj_W, Bhi, Blo);
  sapk_prep_mc <<<3, 256, 0, stream>>>(seg_W, seg_b, proj_W, proj_b, Mc);
  sapk_segscan <<<BATCH, 32, 0, stream>>>(pb, op, startPos, feats);

  dim3 ggrid(MTILES, NTILES / 8);
  sapk_gemm<<<ggrid, 256, 0, stream>>>((const v16bf*)Ahi, (const v16bf*)Alo,
                                       (const v16bf*)Bhi, (const v16bf*)Blo, baseProj);

  sapk_fuse<<<((size_t)BATCH * SEQ * 192) / 256, 256, 0, stream>>>(
      (const float4*)baseProj, feats, (const float4*)Mc, (float4*)d_out);
}
